// ParallelRetention_66803921322506
// MI455X (gfx1250) — compile-verified, matrix operand order checked by reading, not run on hardware
//
#include <hip/hip_runtime.h>
#include <hip/hip_bf16.h>

typedef __attribute__((ext_vector_type(16))) _Float16 v16h;
typedef __attribute__((ext_vector_type(8)))  float    v8f;
typedef __attribute__((ext_vector_type(8)))  _Float16 half8_t;
typedef __attribute__((ext_vector_type(4)))  _Float16 half4_t;
typedef __attribute__((ext_vector_type(4)))  float    float4_t;
typedef __attribute__((ext_vector_type(4)))  unsigned int u32x4;
typedef __attribute__((ext_vector_type(4)))  int      i32x4;
typedef __attribute__((ext_vector_type(8)))  int      i32x8;

#define T_DIM   8192
#define IN_DIM  1024
#define INTER   512
#define OUT_DIM 512

#if __has_builtin(__builtin_amdgcn_tensor_load_to_lds) && \
    __has_builtin(__builtin_amdgcn_s_wait_tensorcnt)
#define HAVE_TDM 1
#else
#define HAVE_TDM 0
#endif

// ---------------------------------------------------------------------------
// WMMA helpers (CDNA5: V_WMMA_F32_16X16X32_F16, D = A(16x32) * B(32x16) + C)
// ---------------------------------------------------------------------------
__device__ __forceinline__ v8f wmma_f16(v16h a, v16h b, v8f c) {
  return __builtin_amdgcn_wmma_f32_16x16x32_f16(false, a, false, b,
                                                (short)0, c, false, false);
}

// Load a 16x32 f16 fragment (A layout, or B stored transposed as [n][k]).
// ISA layout: lanes 0-15 row=lane, halves K0-7 & K16-23; lanes 16-31 same row,
// halves K8-15 & K24-31.  Two 16-byte loads per lane.
__device__ __forceinline__ v16h load_frag(const _Float16* base, int stride,
                                          int lane) {
  const int r    = lane & 15;
  const int koff = (lane >> 4) << 3;           // 0 or 8
  const _Float16* p = base + r * stride + koff;
  union { v16h v; half8_t h[2]; } u;
  u.h[0] = *(const half8_t*)(p);
  u.h[1] = *(const half8_t*)(p + 16);
  return u.v;
}

// ---------------------------------------------------------------------------
// Tensor Data Mover: load a 2D f16 tile [rows x cols] (row stride = cols in
// global, contiguous in LDS).  Issued by one wave; D# built in SGPRs.
// ---------------------------------------------------------------------------
#if HAVE_TDM
__device__ __forceinline__ void tdm_load_tile_f16(unsigned lds_byte_off,
                                                  const _Float16* gsrc,
                                                  unsigned rows, unsigned cols,
                                                  unsigned gstride_elems,
                                                  unsigned tensor_rows) {
  unsigned long long ga = (unsigned long long)(size_t)gsrc;
  u32x4 g0;
  g0[0] = 1u;                                   // count=1, user descriptor
  g0[1] = lds_byte_off;                         // lds_addr
  g0[2] = (unsigned)(ga & 0xffffffffu);         // global_addr[31:0]
  g0[3] = (unsigned)((ga >> 32) & 0x01ffffffu)  // global_addr[56:32]
          | (2u << 30);                         // type = 2 ("image")
  i32x8 g1;
  g1[0] = (int)(1u << 16);                      // data_size=1 (2 bytes)
  g1[1] = (int)((cols & 0xffffu) << 16);        // tensor_dim0[15:0] @ 63:48
  g1[2] = (int)((tensor_rows & 0xffffu) << 16); // tensor_dim1[15:0] @ 95:80
  g1[3] = (int)((cols & 0xffffu) << 16);        // tile_dim0 @ 127:112
  g1[4] = (int)(rows & 0xffffu);                // tile_dim1 @ 143:128
  g1[5] = (int)gstride_elems;                   // tensor_dim0_stride[31:0]
  g1[6] = 0;
  g1[7] = 0;
  i32x4 gz = {0, 0, 0, 0};
#if defined(__clang_major__) && (__clang_major__ >= 23)
  i32x8 gz8 = {0, 0, 0, 0, 0, 0, 0, 0};
  __builtin_amdgcn_tensor_load_to_lds(g0, g1, gz, gz, gz8, 0);
#else
  __builtin_amdgcn_tensor_load_to_lds(g0, g1, gz, gz, 0);
#endif
  __builtin_amdgcn_s_wait_tensorcnt(0);
}
#endif

// ---------------------------------------------------------------------------
// Kernel 0: f32 -> f16 conversion (x and weights), vectorized x4
// ---------------------------------------------------------------------------
__global__ void cvt_f32_f16(const float* __restrict__ src,
                            _Float16* __restrict__ dst, int n) {
  int idx = (blockIdx.x * blockDim.x + threadIdx.x) * 4;
  if (idx < n) {
    float4_t v = *(const float4_t*)(src + idx);
    half4_t o;
    o[0] = (_Float16)v[0]; o[1] = (_Float16)v[1];
    o[2] = (_Float16)v[2]; o[3] = (_Float16)v[3];
    *(half4_t*)(dst + idx) = o;
  }
}

// ---------------------------------------------------------------------------
// Kernel 1: QKV projection.  C[8192x512] = xh[8192x1024] @ W[nxk]^T + bias
// blockIdx.z selects q/k/v.  q,k stored row-major f16; v stored transposed.
// 64x64 block, 8 waves, 2 (16x16) tiles per wave, K-step 32 via LDS.
// ---------------------------------------------------------------------------
__global__ __launch_bounds__(256) void qkv_gemm(
    const _Float16* __restrict__ xh, const _Float16* __restrict__ Wh,
    const float* __restrict__ bq, const float* __restrict__ bk,
    const float* __restrict__ bv,
    _Float16* __restrict__ qout, _Float16* __restrict__ kout,
    _Float16* __restrict__ vT) {
  __shared__ _Float16 As[64][40];
  __shared__ _Float16 Bs[64][40];

  const int mat = blockIdx.z;
  const int m0  = blockIdx.y * 64;
  const int n0  = blockIdx.x * 64;
  const _Float16* W = Wh + (size_t)mat * (INTER * IN_DIM);

  const int tid  = threadIdx.x;
  const int wid  = tid >> 5;
  const int lane = tid & 31;
  const int arow = tid >> 2;            // 64 rows x 4 chunks of 8 halves
  const int acol = (tid & 3) * 8;

  const int t0 = wid, t1 = wid + 8;     // 16 tiles (4x4) over 8 waves
  const int mi0 = t0 >> 2, ni0 = t0 & 3;
  const int mi1 = t1 >> 2, ni1 = t1 & 3;

  v8f acc0 = {}, acc1 = {};

  for (int k = 0; k < IN_DIM; k += 32) {
    *(half8_t*)&As[arow][acol] =
        *(const half8_t*)&xh[(size_t)(m0 + arow) * IN_DIM + k + acol];
    *(half8_t*)&Bs[arow][acol] =
        *(const half8_t*)&W[(size_t)(n0 + arow) * IN_DIM + k + acol];
    __syncthreads();
    {
      v16h a = load_frag(&As[mi0 * 16][0], 40, lane);
      v16h b = load_frag(&Bs[ni0 * 16][0], 40, lane);
      acc0 = wmma_f16(a, b, acc0);
      a = load_frag(&As[mi1 * 16][0], 40, lane);
      b = load_frag(&Bs[ni1 * 16][0], 40, lane);
      acc1 = wmma_f16(a, b, acc1);
    }
    __syncthreads();
  }

  const float* bias = (mat == 0) ? bq : (mat == 1) ? bk : bv;
  const int n  = lane & 15;
  const int mB = (lane >> 4) << 3;      // C layout: VGPR r -> M = mB + r

#pragma unroll
  for (int t = 0; t < 2; ++t) {
    const int mi = t ? mi1 : mi0;
    const int ni = t ? ni1 : ni0;
    const v8f a  = t ? acc1 : acc0;
    const int gn = n0 + ni * 16 + n;
    const float bb = bias[gn];
    if (mat < 2) {
      _Float16* dst = (mat == 0) ? qout : kout;
#pragma unroll
      for (int r = 0; r < 8; ++r)
        dst[(size_t)(m0 + mi * 16 + mB + r) * INTER + gn] =
            (_Float16)(a[r] + bb);
    } else {
      half8_t o = {};
#pragma unroll
      for (int r = 0; r < 8; ++r) o[r] = (_Float16)(a[r] + bb);
      // transposed: vT[n][m..m+7], M contiguous -> one 16B store
      *(half8_t*)&vT[(size_t)gn * T_DIM + m0 + mi * 16 + mB] = o;
    }
  }
}

// ---------------------------------------------------------------------------
// Kernel 2: fused retention.  For 32 rows of h:
//   loop j over T in steps of 32:
//     s[32][32] = d_gamma .* (q_rows @ k[j:j+32]^T)    (waves 0-3, K=512)
//     h[32][512] += s @ v[j:j+32]                       (all 8 waves, K=32)
// q rows staged once into LDS by the Tensor Data Mover (TDM); k and vT are
// pulled from L2-resident global.  Per-block h moments for GroupNorm.
// ---------------------------------------------------------------------------
__global__ __launch_bounds__(256) void retention_fused(
    const _Float16* __restrict__ qh, const _Float16* __restrict__ kh,
    const _Float16* __restrict__ vT, const float* __restrict__ d_gamma,
    _Float16* __restrict__ hh, float* __restrict__ part) {
  __shared__ _Float16 qs[32][INTER];    // 32 KB, persistent q rows (TDM dest)
  __shared__ _Float16 ss[32][40];       // masked scores tile (f16)
  __shared__ float    red[512];

  const int m0   = blockIdx.x * 32;
  const int tid  = threadIdx.x;
  const int wid  = tid >> 5;
  const int lane = tid & 31;
  const int n    = lane & 15;
  const int mB   = (lane >> 4) << 3;

#if HAVE_TDM
  if (wid == 0) {
    // one wave DMAs q[m0..m0+31][0..511] into LDS, waits TENSORcnt==0
    tdm_load_tile_f16((unsigned)(size_t)(void*)&qs[0][0],
                      qh + (size_t)m0 * INTER, 32u, (unsigned)INTER,
                      (unsigned)INTER, (unsigned)T_DIM);
  }
#else
  for (int c = tid; c < 2048; c += 256) {
    int r = c >> 6, col = (c & 63) * 8;
    *(half8_t*)&qs[r][col] =
        *(const half8_t*)&qh[(size_t)(m0 + r) * INTER + col];
  }
#endif

  v8f acc[8];
#pragma unroll
  for (int t = 0; t < 8; ++t) acc[t] = (v8f){};

  const int msub = wid >> 2;            // phase-2 row half (0/1)
  const int nblk = wid & 3;             // phase-2 128-wide column block
  const int sm   = wid >> 1;            // phase-1 score tile row (waves 0-3)
  const int sj   = wid & 1;             // phase-1 score tile col

  __syncthreads();                      // publish q tile

  for (int j = 0; j < T_DIM; j += 32) {
    // stream d_gamma ahead of use
    __builtin_prefetch(&d_gamma[(size_t)(m0 + (tid & 31)) * T_DIM + j + 64],
                       0, 1);
    // ---- phase 1: scores (waves 0-3) ----
    if (wid < 4) {
      v8f sc = {};
#pragma unroll 4
      for (int kk = 0; kk < INTER; kk += 32) {
        v16h a = load_frag(&qs[sm * 16][kk], INTER, lane);
        v16h b = load_frag(&kh[(size_t)(j + sj * 16) * INTER + kk],
                           INTER, lane);
        sc = wmma_f16(a, b, sc);
      }
      const int gj = j + sj * 16 + n;
#pragma unroll
      for (int r = 0; r < 8; ++r) {
        float g = d_gamma[(size_t)(m0 + sm * 16 + mB + r) * T_DIM + gj];
        ss[sm * 16 + mB + r][sj * 16 + n] = (_Float16)(sc[r] * g);
      }
    }
    __syncthreads();
    // ---- phase 2: h += s @ v  (all waves) ----
    {
      v16h a = load_frag(&ss[msub * 16][0], 40, lane);
#pragma unroll
      for (int t = 0; t < 8; ++t) {
        v16h b = load_frag(
            &vT[(size_t)(nblk * 128 + t * 16) * T_DIM + j], T_DIM, lane);
        acc[t] = wmma_f16(a, b, acc[t]);
      }
    }
    __syncthreads();                    // ss reused next iteration
  }

  // epilogue: store h (f16) + partial moments
  float s1 = 0.f, s2 = 0.f;
#pragma unroll
  for (int t = 0; t < 8; ++t) {
    const int gn = nblk * 128 + t * 16 + n;
#pragma unroll
    for (int r = 0; r < 8; ++r) {
      float v = acc[t][r];
      s1 += v; s2 += v * v;
      hh[(size_t)(m0 + msub * 16 + mB + r) * INTER + gn] = (_Float16)v;
    }
  }
  red[tid] = s1; red[256 + tid] = s2;
  __syncthreads();
  for (int s = 128; s > 0; s >>= 1) {
    if (tid < s) { red[tid] += red[tid + s]; red[256 + tid] += red[256 + tid + s]; }
    __syncthreads();
  }
  if (tid == 0) { part[2 * blockIdx.x] = red[0]; part[2 * blockIdx.x + 1] = red[256]; }
}

// ---------------------------------------------------------------------------
// Kernel 3: reduce 256 block partials -> mean, inv_std (deterministic)
// ---------------------------------------------------------------------------
__global__ void finalize_stats(const float* __restrict__ part,
                               float* __restrict__ stats) {
  if (threadIdx.x == 0) {
    double s1 = 0.0, s2 = 0.0;
    for (int i = 0; i < 256; ++i) { s1 += part[2 * i]; s2 += part[2 * i + 1]; }
    const double N = (double)T_DIM * (double)INTER;
    float mean = (float)(s1 / N);
    float var  = (float)(s2 / N) - mean * mean;
    stats[0] = mean;
    stats[1] = rsqrtf(var + 1e-5f);
  }
}

// ---------------------------------------------------------------------------
// Kernel 4: y = PReLU( GN(h) @ Wr^T + br ).  GroupNorm affine applied while
// staging A into LDS; same 64x64 WMMA tiling as kernel 1.  f32 output.
// ---------------------------------------------------------------------------
__global__ __launch_bounds__(256) void out_gemm(
    const _Float16* __restrict__ hh, const _Float16* __restrict__ Wrh,
    const float* __restrict__ gnw, const float* __restrict__ gnb,
    const float* __restrict__ br, const float* __restrict__ prelu,
    const float* __restrict__ stats, float* __restrict__ out) {
  __shared__ _Float16 As[64][40];
  __shared__ _Float16 Bs[64][40];

  const float mean = stats[0];
  const float istd = stats[1];
  const float pw   = prelu[0];

  const int m0 = blockIdx.y * 64;
  const int n0 = blockIdx.x * 64;
  const int tid = threadIdx.x, wid = tid >> 5, lane = tid & 31;
  const int arow = tid >> 2, acol = (tid & 3) * 8;
  const int t0 = wid, t1 = wid + 8;
  const int mi0 = t0 >> 2, ni0 = t0 & 3;
  const int mi1 = t1 >> 2, ni1 = t1 & 3;

  v8f acc0 = {}, acc1 = {};

  for (int k = 0; k < INTER; k += 32) {
    half8_t raw =
        *(const half8_t*)&hh[(size_t)(m0 + arow) * INTER + k + acol];
    half8_t o = {};
#pragma unroll
    for (int e = 0; e < 8; ++e) {
      int c = k + acol + e;
      float f = (float)raw[e];
      o[e] = (_Float16)(((f - mean) * istd) * gnw[c] + gnb[c]);
    }
    *(half8_t*)&As[arow][acol] = o;
    *(half8_t*)&Bs[arow][acol] =
        *(const half8_t*)&Wrh[(size_t)(n0 + arow) * INTER + k + acol];
    __syncthreads();
    {
      v16h a = load_frag(&As[mi0 * 16][0], 40, lane);
      v16h b = load_frag(&Bs[ni0 * 16][0], 40, lane);
      acc0 = wmma_f16(a, b, acc0);
      a = load_frag(&As[mi1 * 16][0], 40, lane);
      b = load_frag(&Bs[ni1 * 16][0], 40, lane);
      acc1 = wmma_f16(a, b, acc1);
    }
    __syncthreads();
  }

  const int n  = lane & 15;
  const int mB = (lane >> 4) << 3;
#pragma unroll
  for (int t = 0; t < 2; ++t) {
    const int mi = t ? mi1 : mi0;
    const int ni = t ? ni1 : ni0;
    const v8f a  = t ? acc1 : acc0;
    const int gn = n0 + ni * 16 + n;
    const float bb = br[gn];
#pragma unroll
    for (int r = 0; r < 8; ++r) {
      float y = a[r] + bb;
      y = (y >= 0.f) ? y : pw * y;
      out[(size_t)(m0 + mi * 16 + mB + r) * OUT_DIM + gn] = y;
    }
  }
}

// ---------------------------------------------------------------------------
// Host-side launch orchestration
// ---------------------------------------------------------------------------
extern "C" void kernel_launch(void* const* d_in, const int* in_sizes, int n_in,
                              void* d_out, int out_size, void* d_ws,
                              size_t ws_size, hipStream_t stream) {
  const float* x      = (const float*)d_in[0];
  const float* dgamma = (const float*)d_in[1];
  const float* Wq     = (const float*)d_in[2];
  const float* bq     = (const float*)d_in[3];
  const float* Wk     = (const float*)d_in[4];
  const float* bk     = (const float*)d_in[5];
  const float* Wv     = (const float*)d_in[6];
  const float* bv     = (const float*)d_in[7];
  const float* Wr     = (const float*)d_in[8];
  const float* br     = (const float*)d_in[9];
  const float* gnw    = (const float*)d_in[10];
  const float* gnb    = (const float*)d_in[11];
  const float* prelu  = (const float*)d_in[12];
  float* out          = (float*)d_out;

  char* ws = (char*)d_ws;
  _Float16* xh  = (_Float16*)(ws);                          // 16 MB
  _Float16* Wqh = (_Float16*)(ws + (size_t)16 * (1 << 20)); // 3x 1MB
  _Float16* Wkh = Wqh + (size_t)INTER * IN_DIM;
  _Float16* Wvh = Wkh + (size_t)INTER * IN_DIM;
  _Float16* Wrh = Wvh + (size_t)INTER * IN_DIM;             // 0.5 MB
  _Float16* qh  = (_Float16*)(ws + (size_t)20 * (1 << 20)); // 4x 8MB
  _Float16* kh  = qh + (size_t)T_DIM * INTER;
  _Float16* vT  = kh + (size_t)T_DIM * INTER;
  _Float16* hh  = vT + (size_t)T_DIM * INTER;
  float* part   = (float*)(ws + (size_t)52 * (1 << 20));    // 256x2 partials
  float* stats  = part + 512;                               // mean, inv_std

  // 1) convert inputs to f16
  {
    int n = T_DIM * IN_DIM;
    cvt_f32_f16<<<n / 4 / 256, 256, 0, stream>>>(x, xh, n);
    n = INTER * IN_DIM;
    cvt_f32_f16<<<n / 4 / 256, 256, 0, stream>>>(Wq, Wqh, n);
    cvt_f32_f16<<<n / 4 / 256, 256, 0, stream>>>(Wk, Wkh, n);
    cvt_f32_f16<<<n / 4 / 256, 256, 0, stream>>>(Wv, Wvh, n);
    n = OUT_DIM * INTER;
    cvt_f32_f16<<<n / 4 / 256, 256, 0, stream>>>(Wr, Wrh, n);
  }
  // 2) q,k,v projections (v transposed)
  qkv_gemm<<<dim3(INTER / 64, T_DIM / 64, 3), 256, 0, stream>>>(
      xh, Wqh, bq, bk, bv, qh, kh, vT);
  // 3) fused retention + moment partials
  retention_fused<<<dim3(T_DIM / 32), 256, 0, stream>>>(qh, kh, vT, dgamma,
                                                        hh, part);
  // 4) GroupNorm scalars
  finalize_stats<<<1, 32, 0, stream>>>(part, stats);
  // 5) normalized output GEMM + PReLU
  out_gemm<<<dim3(OUT_DIM / 64, T_DIM / 64), 256, 0, stream>>>(
      hh, Wrh, gnw, gnb, br, prelu, stats, out);
}
